// CausalMultiHeadAttention_22840636080724
// MI455X (gfx1250) — matrix operand order repaired and checked by reading, not compiled, not run on hardware
//
#include <hip/hip_runtime.h>
#include <cmath>

// ---------------------------------------------------------------------------
// Causal MHA for MI455X (gfx1250): f16 WMMA GEMMs + fp32 flash-attention core.
// ---------------------------------------------------------------------------

typedef _Float16 h4  __attribute__((ext_vector_type(4)));
typedef _Float16 h8  __attribute__((ext_vector_type(8)));
typedef _Float16 h16 __attribute__((ext_vector_type(16)));
typedef float    f4  __attribute__((ext_vector_type(4)));
typedef float    f8  __attribute__((ext_vector_type(8)));

#define DEV __device__ __forceinline__

static constexpr int BB = 4, SS = 2048, DD = 2048, HH = 16, DK = 128;
static constexpr int MROWS = BB * SS;   // 8192
static constexpr int KDIM  = DD;        // 2048
static constexpr int NDIM  = DD;        // 2048

DEV h8 ldh8(const _Float16* p) { return *(const h8*)p; }

DEV h16 mkh16(h8 lo, h8 hi) {
  h16 r;
#pragma unroll
  for (int i = 0; i < 8; ++i) { r[i] = lo[i]; r[i + 8] = hi[i]; }
  return r;
}

DEV f8 wmma16(h16 a, h16 b, f8 c) {
  // D = A(16x32 f16) * B(32x16 f16) + C(16x16 f32)
  return __builtin_amdgcn_wmma_f32_16x16x32_f16(
      /*neg_a=*/false, a, /*neg_b=*/false, b,
      /*c_mod=*/(short)0, c, /*reuse_a=*/false, /*reuse_b=*/false);
}

// ---------------------------------------------------------------------------
// fp32 -> f16 conversion (vectorized by 4)
// ---------------------------------------------------------------------------
__global__ void mha_cvt_f16(const float* __restrict__ in,
                            _Float16* __restrict__ out, int n4) {
  int i = blockIdx.x * blockDim.x + threadIdx.x;
  if (i >= n4) return;
  f4 v = ((const f4*)in)[i];
  h4 o;
#pragma unroll
  for (int j = 0; j < 4; ++j) o[j] = (_Float16)v[j];
  ((h4*)out)[i] = o;
}

// ---------------------------------------------------------------------------
// C[M,N] = A[M,K] * W[N,K]^T.
// Wave computes a 32x64 tile (2 M-tiles x 4 N-tiles of 16x16);
// block = 8 waves (2x4) = 64(M) x 256(N).  K-step 32 -> 8 wmma / 12 b128 loads.
// A fragment reused 4x, B fragment reused 2x within the wave.
// ---------------------------------------------------------------------------
template <bool OUTF16>
__global__ __launch_bounds__(256) void mha_gemm_xwT(
    const _Float16* __restrict__ A, const _Float16* __restrict__ W,
    _Float16* __restrict__ Ch, float* __restrict__ Cf) {
  const int lane = threadIdx.x & 31;
  const int wv   = threadIdx.x >> 5;
  const int half = lane >> 4;
  const int l16  = lane & 15;
  const int m0 = blockIdx.y * 64 + (wv & 1) * 32;
  const int n0 = blockIdx.x * 256 + (wv >> 1) * 64;

  f8 acc[2][4] = {};
  for (int k = 0; k < KDIM; k += 32) {
    // A fragments: lane(row m0+l16+16*mt) holds K {half*8..+7} U {16+half*8..+7}
    const _Float16* pa0 = A + (size_t)(m0 + l16) * KDIM + k + half * 8;
    const h16 a0 = mkh16(ldh8(pa0), ldh8(pa0 + 16));
    const _Float16* pa1 = pa0 + (size_t)16 * KDIM;
    const h16 a1 = mkh16(ldh8(pa1), ldh8(pa1 + 16));
    // B fragments: lane(col n0+l16+16*nt) holds K {half*8..+7} U {16+half*8..+7}
#pragma unroll
    for (int nt = 0; nt < 4; ++nt) {
      const _Float16* pw = W + (size_t)(n0 + nt * 16 + l16) * KDIM + k + half * 8;
      const h16 b = mkh16(ldh8(pw), ldh8(pw + 16));
      acc[0][nt] = wmma16(a0, b, acc[0][nt]);
      acc[1][nt] = wmma16(a1, b, acc[1][nt]);
    }
  }
#pragma unroll
  for (int mt = 0; mt < 2; ++mt)
#pragma unroll
    for (int nt = 0; nt < 4; ++nt) {
      const int col = n0 + nt * 16 + l16;
#pragma unroll
      for (int j = 0; j < 8; ++j) {
        const int row = m0 + mt * 16 + j + 8 * half;
        if constexpr (OUTF16)
          Ch[(size_t)row * NDIM + col] = (_Float16)acc[mt][nt][j];
        else
          Cf[(size_t)row * NDIM + col] = acc[mt][nt][j];
      }
    }
}

// ---------------------------------------------------------------------------
// RoPE (interleaved pairs) + head permute: [B,S,H,DK] f16 -> [B,H,S,DK] f16
// ---------------------------------------------------------------------------
__global__ void mha_rope_permute(const _Float16* __restrict__ in,
                                 _Float16* __restrict__ out) {
  int p = blockIdx.x * blockDim.x + threadIdx.x;  // pair id, c2 fastest
  const int c2 = p & 63;
  int t = p >> 6;
  const int h = t & 15; t >>= 4;
  const int s = t & (SS - 1);
  const int b = t >> 11;
  const size_t ii = ((size_t)(b * SS + s) * DD) + h * DK + 2 * c2;
  const float x0 = (float)in[ii], x1 = (float)in[ii + 1];
  // inv_freq = 10000^(-2*c2/128) = 2^(-c2 * (2/128) * log2(10000))
  const float inv = exp2f(-(float)c2 * (2.0f / DK) * 13.287712379549449f);
  float sn, cs;
  sincosf((float)s * inv, &sn, &cs);
  const size_t oi = (((size_t)(b * HH + h) * SS + s) * DK) + 2 * c2;
  out[oi]     = (_Float16)(x0 * cs - x1 * sn);
  out[oi + 1] = (_Float16)(x1 * cs + x0 * sn);
}

// ---------------------------------------------------------------------------
// V transpose: [B,S,H,DK] f16 -> [B,H,DK,S] f16 (LDS 32x32 tiles, coalesced)
// ---------------------------------------------------------------------------
__global__ void mha_transpose_v(const _Float16* __restrict__ in,
                                _Float16* __restrict__ out) {
  __shared__ _Float16 tile[32][33];
  const int bh = blockIdx.z, b = bh >> 4, h = bh & 15;
  const int s0 = blockIdx.x * 32, c0 = blockIdx.y * 32;
  const int tx = threadIdx.x & 31, ty = threadIdx.x >> 5;  // ty 0..7
#pragma unroll
  for (int j = 0; j < 4; ++j) {
    const int r = ty + j * 8;
    tile[r][tx] = in[(size_t)(b * SS + s0 + r) * DD + h * DK + c0 + tx];
  }
  __syncthreads();
#pragma unroll
  for (int j = 0; j < 4; ++j) {
    const int r = ty + j * 8;
    out[((size_t)bh * DK + c0 + r) * SS + s0 + tx] = tile[tx][r];
  }
}

// ---------------------------------------------------------------------------
// Flash attention: wave = 16 query rows, KV chunks of 32, online softmax.
// Q,K: [B,H,S,DK] f16 (RoPE'd);  Vt: [B,H,DK,S] f16;  Oh: [B,S,D] f16.
// ---------------------------------------------------------------------------
__global__ __launch_bounds__(256) void mha_flash_attn(
    const _Float16* __restrict__ Q, const _Float16* __restrict__ Km,
    const _Float16* __restrict__ Vt, _Float16* __restrict__ Oh) {
  __shared__ alignas(16) _Float16 ldsP[8][16][32];  // per-wave P staging
  const int lane = threadIdx.x & 31, wv = threadIdx.x >> 5;
  const int half = lane >> 4, l16 = lane & 15;
  const int bh = blockIdx.y, b = bh >> 4, h = bh & 15;
  const int q0 = blockIdx.x * 128 + wv * 16;

  const _Float16* Qb = Q  + (size_t)bh * SS * DK;
  const _Float16* Kb = Km + (size_t)bh * SS * DK;
  const _Float16* Vb = Vt + (size_t)bh * DK * SS;

  // Q A-fragments for dk chunks of 32 (kept in registers for the whole tile)
  h16 aQ[4];
#pragma unroll
  for (int kt = 0; kt < 4; ++kt) {
    const _Float16* p = Qb + (size_t)(q0 + l16) * DK + kt * 32 + half * 8;
    aQ[kt] = mkh16(ldh8(p), ldh8(p + 16));
  }

  f8 accO[8] = {};            // 16 x 128 output accumulator (8 dk-tiles)
  float m[8], l[8];
#pragma unroll
  for (int j = 0; j < 8; ++j) { m[j] = -3.0e38f; l[j] = 0.0f; }

  constexpr float CEXP = 0.12751743f;  // (1/sqrt(128)) * log2(e)

  for (int kv0 = 0; kv0 < q0 + 16; kv0 += 32) {
    // ---- S = Q * K^T for 32 keys (two 16x16 tiles, K-dim = 128 = 4 wmma) ---
    f8 s0 = {}, s1 = {};
#pragma unroll
    for (int kt = 0; kt < 4; ++kt) {
      const _Float16* p0 = Kb + (size_t)(kv0 + l16) * DK + kt * 32 + half * 8;
      s0 = wmma16(aQ[kt], mkh16(ldh8(p0), ldh8(p0 + 16)), s0);
      const _Float16* p1 = p0 + (size_t)16 * DK;
      s1 = wmma16(aQ[kt], mkh16(ldh8(p1), ldh8(p1 + 16)), s1);
    }
    // ---- causal mask (diagonal chunks only) ----
    if (kv0 + 31 > q0) {
#pragma unroll
      for (int j = 0; j < 8; ++j) {
        const int row = q0 + j + 8 * half;
        if (kv0 + l16 > row)      s0[j] = -3.0e38f;
        if (kv0 + 16 + l16 > row) s1[j] = -3.0e38f;
      }
    }
    // ---- online softmax (fp32, row stats live in the 16-lane half) ----
    float sc[8];
#pragma unroll
    for (int j = 0; j < 8; ++j) {
      float t = fmaxf(s0[j], s1[j]);
      t = fmaxf(t, __shfl_xor(t, 1, 32));
      t = fmaxf(t, __shfl_xor(t, 2, 32));
      t = fmaxf(t, __shfl_xor(t, 4, 32));
      t = fmaxf(t, __shfl_xor(t, 8, 32));
      const float mn = fmaxf(m[j], t);
      sc[j] = exp2f((m[j] - mn) * CEXP);
      m[j]  = mn;
      const float p0 = exp2f((s0[j] - mn) * CEXP);
      const float p1 = exp2f((s1[j] - mn) * CEXP);
      s0[j] = p0; s1[j] = p1;
      float rs = p0 + p1;
      rs += __shfl_xor(rs, 1, 32);
      rs += __shfl_xor(rs, 2, 32);
      rs += __shfl_xor(rs, 4, 32);
      rs += __shfl_xor(rs, 8, 32);
      l[j] = l[j] * sc[j] + rs;
    }
#pragma unroll
    for (int nt = 0; nt < 8; ++nt)
#pragma unroll
      for (int j = 0; j < 8; ++j) accO[nt][j] *= sc[j];

    // ---- P: C-layout -> A-layout via per-wave LDS tile (16x32 f16) ----
#pragma unroll
    for (int j = 0; j < 8; ++j) {
      ldsP[wv][j + 8 * half][l16]      = (_Float16)s0[j];
      ldsP[wv][j + 8 * half][16 + l16] = (_Float16)s1[j];
    }
    asm volatile("s_wait_dscnt 0" ::: "memory");
    const h8 lo = *(const h8*)&ldsP[wv][l16][half * 8];
    const h8 hi = *(const h8*)&ldsP[wv][l16][16 + half * 8];
    const h16 aP = mkh16(lo, hi);

    // ---- O += P * V  (B frags are contiguous rows of Vt) ----
#pragma unroll
    for (int nt = 0; nt < 8; ++nt) {
      const _Float16* pv = Vb + (size_t)(nt * 16 + l16) * SS + kv0 + half * 8;
      accO[nt] = wmma16(aP, mkh16(ldh8(pv), ldh8(pv + 16)), accO[nt]);
    }
  }

  // ---- epilogue: O / l  ->  attn [B,S,D] f16 ----
#pragma unroll
  for (int nt = 0; nt < 8; ++nt) {
    const int c = nt * 16 + l16;
#pragma unroll
    for (int j = 0; j < 8; ++j) {
      const int s = q0 + j + 8 * half;
      Oh[(size_t)(b * SS + s) * DD + h * DK + c] = (_Float16)(accO[nt][j] / l[j]);
    }
  }
}

// ---------------------------------------------------------------------------
extern "C" void kernel_launch(void* const* d_in, const int* in_sizes, int n_in,
                              void* d_out, int out_size, void* d_ws, size_t ws_size,
                              hipStream_t stream) {
  const float* x  = (const float*)d_in[0];
  const float* Wq = (const float*)d_in[1];
  const float* Wk = (const float*)d_in[2];
  const float* Wv = (const float*)d_in[3];
  const float* Wo = (const float*)d_in[4];
  float* out = (float*)d_out;

  // workspace layout (288 MiB total)
  const size_t SZ_W = (size_t)DD * DD * sizeof(_Float16);       // 8 MiB
  const size_t SZ_X = (size_t)BB * SS * DD * sizeof(_Float16);  // 32 MiB
  char* ws = (char*)d_ws;
  size_t off = 0;
  auto take = [&](size_t bytes) { char* p = ws + off; off += bytes; return p; };
  _Float16* WqH = (_Float16*)take(SZ_W);
  _Float16* WkH = (_Float16*)take(SZ_W);
  _Float16* WvH = (_Float16*)take(SZ_W);
  _Float16* WoH = (_Float16*)take(SZ_W);
  _Float16* xh  = (_Float16*)take(SZ_X);
  _Float16* Qp  = (_Float16*)take(SZ_X);
  _Float16* Kp  = (_Float16*)take(SZ_X);
  _Float16* Vp  = (_Float16*)take(SZ_X);
  _Float16* Qh  = (_Float16*)take(SZ_X);
  _Float16* Kh  = (_Float16*)take(SZ_X);
  _Float16* Vtr = (_Float16*)take(SZ_X);
  _Float16* Ah  = (_Float16*)take(SZ_X);

  const int nX4 = BB * SS * DD / 4;   // 4,194,304
  const int nW4 = DD * DD / 4;        // 1,048,576
  mha_cvt_f16<<<nX4 / 256, 256, 0, stream>>>(x,  xh,  nX4);
  mha_cvt_f16<<<nW4 / 256, 256, 0, stream>>>(Wq, WqH, nW4);
  mha_cvt_f16<<<nW4 / 256, 256, 0, stream>>>(Wk, WkH, nW4);
  mha_cvt_f16<<<nW4 / 256, 256, 0, stream>>>(Wv, WvH, nW4);
  mha_cvt_f16<<<nW4 / 256, 256, 0, stream>>>(Wo, WoH, nW4);

  const dim3 gemmGrid(NDIM / 256, MROWS / 64);  // (8, 128)
  mha_gemm_xwT<true><<<gemmGrid, 256, 0, stream>>>(xh, WqH, Qp, nullptr);
  mha_gemm_xwT<true><<<gemmGrid, 256, 0, stream>>>(xh, WkH, Kp, nullptr);
  mha_gemm_xwT<true><<<gemmGrid, 256, 0, stream>>>(xh, WvH, Vp, nullptr);

  const int pairs = BB * SS * DD / 2;  // 8,388,608
  mha_rope_permute<<<pairs / 256, 256, 0, stream>>>(Qp, Qh);
  mha_rope_permute<<<pairs / 256, 256, 0, stream>>>(Kp, Kh);
  mha_transpose_v<<<dim3(SS / 32, DK / 32, BB * HH), 256, 0, stream>>>(Vp, Vtr);

  mha_flash_attn<<<dim3(SS / 128, BB * HH), 256, 0, stream>>>(Qh, Kh, Vtr, Ah);

  mha_gemm_xwT<false><<<gemmGrid, 256, 0, stream>>>(Ah, WoH, nullptr, out);
}
